// RandomNetwork_30382598652290
// MI455X (gfx1250) — matrix operand order repaired and checked by reading, not compiled, not run on hardware
//
#include <hip/hip_runtime.h>
#include <cstdint>
#include <cstddef>
#include <random>
#include <algorithm>

// ---------------------------------------------------------------------------
// Types for WMMA fragments (wave32, gfx1250)
// ---------------------------------------------------------------------------
typedef __attribute__((ext_vector_type(16))) _Float16 v16h;
typedef __attribute__((ext_vector_type(8)))  _Float16 v8h;
typedef __attribute__((ext_vector_type(8)))  float    v8f;

#define N_NODES 24
#define C_IN    64
#define CCH     128
#define BATCH   16
#define H_IN    64
#define H_OUT   32
#define MPIX    (BATCH * H_OUT * H_OUT)   // 16384 pixels per node output

// ---------------------------------------------------------------------------
// Node 0: gate + ReLU, NCHW -> [pixel][channel] layout (64x64 spatial, 64 ch)
// ---------------------------------------------------------------------------
__global__ void agg_input_kernel(const float* __restrict__ x,
                                 const float* __restrict__ agg_w,
                                 float* __restrict__ Z0) {
  int t = blockIdx.x * blockDim.x + threadIdx.x;        // over 16*64*64*64
  int c  = t & 63;
  int p  = t >> 6;                                      // b*4096 + y*64 + x
  int xx = p & 63;
  int y  = (p >> 6) & 63;
  int b  = p >> 12;
  float g = 1.0f / (1.0f + __expf(-agg_w[0]));
  float v = x[(((size_t)(b * C_IN + c) * H_IN + y) * H_IN + xx)] * g;
  Z0[t] = v > 0.0f ? v : 0.0f;
}

// ---------------------------------------------------------------------------
// Node 0 depthwise 3x3 stride-2 (SAME: pad_lo=0, pad_hi=1), f32 -> f16 A-matrix
// ---------------------------------------------------------------------------
__global__ void dw0_kernel(const float* __restrict__ Z0,
                           const float* __restrict__ dw0,
                           _Float16* __restrict__ A0) {
  int t = blockIdx.x * blockDim.x + threadIdx.x;        // over 16384*64
  int c  = t & 63;
  int p  = t >> 6;                                      // b*1024 + y*32 + x
  int xx = p & 31;
  int y  = (p >> 5) & 31;
  int b  = p >> 10;
  float s = 0.0f;
#pragma unroll
  for (int ky = 0; ky < 3; ++ky) {
    int iy = 2 * y + ky;
    if (iy >= H_IN) continue;
#pragma unroll
    for (int kx = 0; kx < 3; ++kx) {
      int ix = 2 * xx + kx;
      if (ix >= H_IN) continue;
      s += dw0[(ky * 3 + kx) * C_IN + c] *
           Z0[((size_t)(b * 4096 + iy * H_IN + ix)) * C_IN + c];
    }
  }
  A0[t] = (_Float16)s;
}

// ---------------------------------------------------------------------------
// Nodes 1..23: gated aggregation of up to 4 predecessors + ReLU
// ---------------------------------------------------------------------------
__global__ void agg_kernel(const float* __restrict__ p0, const float* __restrict__ p1,
                           const float* __restrict__ p2, const float* __restrict__ p3,
                           int k, int node,
                           const float* __restrict__ agg_w,
                           float* __restrict__ Z) {
  int t = blockIdx.x * blockDim.x + threadIdx.x;        // over MPIX*128
  float s = (1.0f / (1.0f + __expf(-agg_w[node * 4 + 0]))) * p0[t];
  if (k > 1) s += (1.0f / (1.0f + __expf(-agg_w[node * 4 + 1]))) * p1[t];
  if (k > 2) s += (1.0f / (1.0f + __expf(-agg_w[node * 4 + 2]))) * p2[t];
  if (k > 3) s += (1.0f / (1.0f + __expf(-agg_w[node * 4 + 3]))) * p3[t];
  Z[t] = s > 0.0f ? s : 0.0f;
}

// ---------------------------------------------------------------------------
// Nodes 1..23 depthwise 3x3 stride-1 (SAME: pad 1), f32 -> f16 A-matrix
// ---------------------------------------------------------------------------
__global__ void dw_kernel(const float* __restrict__ Z,
                          const float* __restrict__ dwn,
                          _Float16* __restrict__ A) {
  int t = blockIdx.x * blockDim.x + threadIdx.x;        // over MPIX*128
  int c  = t & 127;
  int p  = t >> 7;
  int xx = p & 31;
  int y  = (p >> 5) & 31;
  int b  = p >> 10;
  float s = 0.0f;
#pragma unroll
  for (int ky = 0; ky < 3; ++ky) {
    int iy = y + ky - 1;
    if ((unsigned)iy >= (unsigned)H_OUT) continue;
#pragma unroll
    for (int kx = 0; kx < 3; ++kx) {
      int ix = xx + kx - 1;
      if ((unsigned)ix >= (unsigned)H_OUT) continue;
      s += dwn[(ky * 3 + kx) * CCH + c] *
           Z[((size_t)((b << 10) + (iy << 5) + ix)) * CCH + c];
    }
  }
  A[t] = (_Float16)s;
}

// ---------------------------------------------------------------------------
// Pack a KxN (N=128) f32 weight matrix into WMMA B-fragment order (f16).
// B order packed here (16-bit, 32x16 per wave32; contiguous k, not the ISA table order): lane l holds column n=l&15;
// lanes 0-15 hold K=0..15 of the k-step, lanes 16-31 hold K=16..31.
// Fragment half j (=VGPR v*2+pos) corresponds to k = s*32 + (l>>4)*16 + j,
// so each lane's 16 halves are K-contiguous -> 32B vector load in the GEMM.
// dst index: (((tile * ksteps + s) * 32 + lane) * 16 + j)
// ---------------------------------------------------------------------------
__global__ void pack_b_kernel(const float* __restrict__ src,  // K x 128 row-major
                              _Float16* __restrict__ dst, int K) {
  int ksteps = K >> 5;
  int total  = 8 * ksteps * 32 * 16;
  int t = blockIdx.x * blockDim.x + threadIdx.x;
  if (t >= total) return;
  int j    = t & 15;
  int l    = (t >> 4) & 31;
  int rest = t >> 9;                 // tile*ksteps + s
  int s    = rest % ksteps;
  int tile = rest / ksteps;
  int half = l >> 4;
  int n    = tile * 16 + (l & 15);
  int k    = s * 32 + half * 16 + j;
  dst[t] = (_Float16)src[(size_t)k * CCH + n];
}

// ---------------------------------------------------------------------------
// GEMM: out[M x 128] = A[M x K] * B[K x 128], f16 in / f32 acc, fused affine.
// Block = 256 threads = 8 waves; wave w -> N-tile w; block -> one 16-row M tile.
// A layout (16-bit 16x32): lane l holds row m=l&15;
//   order used here: VGPR v (<4): k = half*16 + v*2+pos ; VGPR v(>=4): k = half*16 + 8 + (v-4)*2+pos
// -> halves 0..7 and 8..15 are each 8 K-contiguous halves => two b128 loads.
// ---------------------------------------------------------------------------
__global__ void gemm_wmma_kernel(const _Float16* __restrict__ A,
                                 const _Float16* __restrict__ Bp,
                                 const float* __restrict__ gamma,
                                 const float* __restrict__ beta,
                                 float* __restrict__ out, int K) {
  const int lane  = threadIdx.x & 31;
  const int wave  = threadIdx.x >> 5;          // N tile 0..7
  const int m0    = blockIdx.x << 4;           // M tile
  const int half  = lane >> 4;
  const int mrow  = m0 + (lane & 15);
  const int kstep = K >> 5;

  v8f acc = {};
  union AFrag { v16h v; v8h p[2]; } a;
  const _Float16* arow = A + (size_t)mrow * K;

  for (int s = 0; s < kstep; ++s) {
    const int kk = s << 5;
    a.p[0] = *(const v8h*)(arow + kk + half * 16);       // k = kk+half*16 .. +7
    a.p[1] = *(const v8h*)(arow + kk + half * 16 + 8);   // k = kk+half*16+8 .. +15
    v16h b = *(const v16h*)(Bp + (((size_t)wave * kstep + s) * 32 + lane) * 16);
    acc = __builtin_amdgcn_wmma_f32_16x16x32_f16(
        /*neg_a=*/false, a.v, /*neg_b=*/false, b,
        /*c_mod=*/(short)0, acc, /*reuse_a=*/false, /*reuse_b=*/false);
  }

  const int col = (wave << 4) + (lane & 15);
  const float gm = gamma[col];
  const float bt = beta[col];
#pragma unroll
  for (int r = 0; r < 8; ++r) {
    int m = m0 + r + (half << 3);              // C/D: VGPR r -> M = r + 8*half
    out[(size_t)m * CCH + col] = acc[r] * gm + bt;
  }
}

// ---------------------------------------------------------------------------
// Final: mean of sink-node outputs, [pixel][channel] -> NCHW f32
// ---------------------------------------------------------------------------
__global__ void final_mean_kernel(const float* __restrict__ nodeBuf,
                                  unsigned sink_mask, float inv_ns,
                                  float* __restrict__ out) {
  int t = blockIdx.x * blockDim.x + threadIdx.x;   // over 16*128*32*32
  int xx = t & 31;
  int y  = (t >> 5) & 31;
  int c  = (t >> 10) & 127;
  int b  = t >> 17;
  size_t p = ((size_t)b << 10) + (y << 5) + xx;
  float s = 0.0f;
  unsigned m = sink_mask;
  while (m) {
    int i = __ffs(m) - 1;
    m &= m - 1;
    s += nodeBuf[(size_t)i * MPIX * CCH + p * CCH + c];
  }
  out[t] = s * inv_ns;
}

// ---------------------------------------------------------------------------
// Host-side: reproduce np.random.RandomState(42) DAG (legacy MT19937 streams)
// ---------------------------------------------------------------------------
static void build_dag(int preds[N_NODES][4], int* ndeg,
                      unsigned* sink_mask, int* nsinks) {
  std::mt19937 mt(42u);  // same init_genrand seeding as numpy RandomState
  auto next32 = [&]() -> uint64_t { return (uint64_t)mt(); };
  auto next64 = [&]() -> uint64_t {
    uint64_t hi = next32(); uint64_t lo = next32(); return (hi << 32) | lo;
  };
  // legacy randint (int64 path): masked rejection on 64-bit draws
  auto randint = [&](long long low, long long highExcl) -> long long {
    uint64_t rng = (uint64_t)(highExcl - 1 - low);
    if (rng == 0) return low;
    uint64_t mask = rng;
    mask |= mask >> 1; mask |= mask >> 2; mask |= mask >> 4;
    mask |= mask >> 8; mask |= mask >> 16; mask |= mask >> 32;
    uint64_t v;
    do { v = next64() & mask; } while (v > rng);
    return low + (long long)v;
  };
  // random_interval (max inclusive, <=32-bit path): masked rejection on 32-bit
  auto interval = [&](uint64_t mx) -> uint64_t {
    if (!mx) return 0;
    uint64_t mask = mx;
    mask |= mask >> 1; mask |= mask >> 2; mask |= mask >> 4;
    mask |= mask >> 8; mask |= mask >> 16; mask |= mask >> 32;
    uint64_t v;
    do { v = next32() & mask; } while (v > mx);
    return v;
  };

  bool has_succ[N_NODES] = {false};
  ndeg[0] = 0;
  for (int i = 1; i < N_NODES; ++i) {
    int mk = i < 4 ? i : 4;
    int k = (int)randint(1, mk + 1);
    int perm[N_NODES];
    for (int q = 0; q < i; ++q) perm[q] = q;
    for (int q = i - 1; q > 0; --q) {        // Fisher-Yates (numpy shuffle)
      int j = (int)interval((uint64_t)q);
      int tmp = perm[q]; perm[q] = perm[j]; perm[j] = tmp;
    }
    std::sort(perm, perm + k);               // reference sorts chosen preds
    ndeg[i] = k;
    for (int q = 0; q < k; ++q) { preds[i][q] = perm[q]; has_succ[perm[q]] = true; }
    for (int q = k; q < 4; ++q) preds[i][q] = perm[0];
  }
  unsigned sm = 0; int ns = 0;
  for (int i = 0; i < N_NODES; ++i)
    if (!has_succ[i]) { sm |= 1u << i; ++ns; }
  *sink_mask = sm; *nsinks = ns;
}

// ---------------------------------------------------------------------------
// kernel_launch
// ---------------------------------------------------------------------------
extern "C" void kernel_launch(void* const* d_in, const int* in_sizes, int n_in,
                              void* d_out, int out_size, void* d_ws, size_t ws_size,
                              hipStream_t stream) {
  const float* x     = (const float*)d_in[0];
  const float* agg_w = (const float*)d_in[1];
  const float* dw0   = (const float*)d_in[2];
  const float* pw0   = (const float*)d_in[3];
  const float* dw    = (const float*)d_in[4];
  const float* pw    = (const float*)d_in[5];
  const float* gamma = (const float*)d_in[6];
  const float* beta  = (const float*)d_in[7];

  int preds[N_NODES][4];
  int ndeg[N_NODES];
  unsigned sink_mask; int nsinks;
  build_dag(preds, ndeg, &sink_mask, &nsinks);

  // Workspace carve (~223 MB): 24 node outputs, agg scratch, A f16, packed B f16
  float*    nodeBuf = (float*)d_ws;                              // 24 * MPIX*128 f32
  float*    Z       = nodeBuf + (size_t)N_NODES * MPIX * CCH;    // 16*64*64*64 f32 (max)
  _Float16* Ahalf   = (_Float16*)(Z + (size_t)BATCH * H_IN * H_IN * C_IN);
  _Float16* BpackN  = Ahalf + (size_t)MPIX * CCH;                // 23 * 16384 halves
  _Float16* Bpack0  = BpackN + 23ull * 16384;                    // 8192 halves

  // Pack pointwise weights into WMMA B-fragment order
  for (int j = 0; j < 23; ++j)
    pack_b_kernel<<<16384 / 256, 256, 0, stream>>>(pw + (size_t)j * CCH * CCH,
                                                   BpackN + (size_t)j * 16384, CCH);
  pack_b_kernel<<<8192 / 256, 256, 0, stream>>>(pw0, Bpack0, C_IN);

  // Node 0: gate+relu -> depthwise s2 -> WMMA pointwise (K=64) + affine
  agg_input_kernel<<<(BATCH * H_IN * H_IN * C_IN) / 256, 256, 0, stream>>>(x, agg_w, Z);
  dw0_kernel<<<(MPIX * C_IN) / 256, 256, 0, stream>>>(Z, dw0, Ahalf);
  gemm_wmma_kernel<<<MPIX / 16, 256, 0, stream>>>(Ahalf, Bpack0, gamma, beta,
                                                  nodeBuf, C_IN);

  // Nodes 1..23
  for (int i = 1; i < N_NODES; ++i) {
    const float* p0 = nodeBuf + (size_t)preds[i][0] * MPIX * CCH;
    const float* p1 = nodeBuf + (size_t)preds[i][1] * MPIX * CCH;
    const float* p2 = nodeBuf + (size_t)preds[i][2] * MPIX * CCH;
    const float* p3 = nodeBuf + (size_t)preds[i][3] * MPIX * CCH;
    agg_kernel<<<(MPIX * CCH) / 256, 256, 0, stream>>>(p0, p1, p2, p3, ndeg[i], i,
                                                       agg_w, Z);
    dw_kernel<<<(MPIX * CCH) / 256, 256, 0, stream>>>(Z, dw + (size_t)(i - 1) * 9 * CCH,
                                                      Ahalf);
    gemm_wmma_kernel<<<MPIX / 16, 256, 0, stream>>>(Ahalf, BpackN + (size_t)(i - 1) * 16384,
                                                    gamma + (size_t)i * CCH,
                                                    beta + (size_t)i * CCH,
                                                    nodeBuf + (size_t)i * MPIX * CCH, CCH);
  }

  // Mean over sink nodes -> NCHW f32 output
  final_mean_kernel<<<(BATCH * CCH * H_OUT * H_OUT) / 256, 256, 0, stream>>>(
      nodeBuf, sink_mask, 1.0f / (float)nsinks, (float*)d_out);
}
